// MultiplexedFinalRankerMMoE_23218593202346
// MI455X (gfx1250) — compile-verified, matrix operand order checked by reading, not run on hardware
//
#include <hip/hip_runtime.h>
#include <hip/hip_bf16.h>
#include <cstdint>

#define B_   8192
#define IN_  1024
#define E_   16
#define H1_  1024
#define H2_  512
#define T_   2
#define HD1_ 512
#define HD2_ 256

#define USE_ASYNC 1  // flip to 0 if global_load_async_to_lds_b128 is rejected

typedef __attribute__((ext_vector_type(16))) __bf16 v16bf;
typedef __attribute__((ext_vector_type(8)))  __bf16 v8bf;
typedef __attribute__((ext_vector_type(8)))  float  v8f;

__device__ __forceinline__ unsigned short f2bf(float f) {
  unsigned int u = __float_as_uint(f);
  unsigned int r = (u + 0x7FFFu + ((u >> 16) & 1u)) >> 16;  // RNE
  return (unsigned short)r;
}
__device__ __forceinline__ float bf2f(unsigned short u) {
  return __uint_as_float(((unsigned int)u) << 16);
}

#if USE_ASYNC
// CDNA5 async copy: LDS[vdst] = MEM[vaddr], tracked by ASYNCcnt.
__device__ __forceinline__ void async_b128(unsigned lds_off,
                                           const void* gaddr) {
  asm volatile("global_load_async_to_lds_b128 %0, %1, off" ::"v"(lds_off),
               "v"(gaddr)
               : "memory");
}
__device__ __forceinline__ void wait_async0() {
  asm volatile("s_wait_asynccnt 0x0" ::: "memory");
}
#endif

// ---------------------------------------------------------------------------
// f32 -> bf16 straight conversion (x)
// ---------------------------------------------------------------------------
__global__ void cvt_f32_bf16(const float* __restrict__ in,
                             unsigned short* __restrict__ out, int n) {
  int i = (blockIdx.x * blockDim.x + threadIdx.x) * 4;
  if (i + 3 < n) {
    float4 v = *(const float4*)(in + i);
    out[i + 0] = f2bf(v.x);
    out[i + 1] = f2bf(v.y);
    out[i + 2] = f2bf(v.z);
    out[i + 3] = f2bf(v.w);
  }
}

// ---------------------------------------------------------------------------
// f32 [Z][R][C] -> bf16 [Z][C][R]  (weight transpose so B is K-contiguous)
// grid (C/32, R/32, Z), block (32,8)
// ---------------------------------------------------------------------------
__global__ void transpose_f32_bf16(const float* __restrict__ in,
                                   unsigned short* __restrict__ out,
                                   int R, int C) {
  __shared__ float tile[32][33];
  int z = blockIdx.z;
  const float* inz = in + (size_t)z * R * C;
  unsigned short* outz = out + (size_t)z * R * C;
  int c0 = blockIdx.x * 32, r0 = blockIdx.y * 32;
  int tx = threadIdx.x, ty = threadIdx.y;
#pragma unroll
  for (int j = 0; j < 32; j += 8)
    tile[ty + j][tx] = inz[(size_t)(r0 + ty + j) * C + c0 + tx];
  __syncthreads();
#pragma unroll
  for (int j = 0; j < 32; j += 8)
    outz[(size_t)(c0 + ty + j) * R + r0 + tx] = f2bf(tile[tx][ty + j]);
}

// ---------------------------------------------------------------------------
// WMMA bf16 GEMM:  C[z] = act(A[z](MxK) * Bt[z](NxK)^T + bias[z])   (bf16 out)
// Block tile 128x256x32, 256 threads = 8 waves (2x4), each wave 64x64
// (4x4 wmma tiles -> 16 v_wmma per k-step, 16 ds_load_b128 per k-step).
// Double-buffered LDS + async global->LDS copies (ASYNCcnt).
// ---------------------------------------------------------------------------
#define BM 128
#define BN 256
#define BK 32
#define STR 40  // padded LDS row stride (elements) -> conflict-free frag loads

union Frag16 { v16bf v; v8bf h[2]; };

template <bool RELU>
__global__ __launch_bounds__(256) void wmma_gemm_bf16(
    const unsigned short* __restrict__ A, long long sAz,
    const unsigned short* __restrict__ Bt, long long sBz,
    const float* __restrict__ bias, long long sbz,
    unsigned short* __restrict__ C, long long sCz,
    int M, int N, int K, int lda, int ldb, int ldc) {
  __shared__ __align__(16) unsigned short As[2 * BM * STR];
  __shared__ __align__(16) unsigned short Bs[2 * BN * STR];

  int z = blockIdx.z;
  const unsigned short* Az = A + (size_t)z * sAz;
  const unsigned short* Bz = Bt + (size_t)z * sBz;
  const float* biasz = bias + (size_t)z * sbz;
  unsigned short* Cz = C + (size_t)z * sCz;

  int n_blk = blockIdx.x * BN;
  int m_blk = blockIdx.y * BM;
  int tid = threadIdx.x;
  int lane = tid & 31;
  int wid = tid >> 5;
  int wm = wid & 1;   // 0..1 -> 64-row half of BM
  int wn = wid >> 1;  // 0..3 -> 64-col quarter of BN
  int l16 = lane & 15;
  int half = lane >> 4;

  // per-thread staging geometry: 16B chunk = 8 bf16
  int rowA = tid >> 2;        // 0..63
  int ko = (tid & 3) * 8;     // 0,8,16,24
  const unsigned short* ga = Az + (size_t)(m_blk + rowA) * lda + ko;
  const unsigned short* gb = Bz + (size_t)(n_blk + rowA) * ldb + ko;
  unsigned asBase = (unsigned)(size_t)(&As[0]) + (unsigned)(rowA * STR + ko) * 2;
  unsigned bsBase = (unsigned)(size_t)(&Bs[0]) + (unsigned)(rowA * STR + ko) * 2;

  auto issue_tile = [&](int buf, int kt) {
    const unsigned short* gaa = ga + kt * BK;
    const unsigned short* gbb = gb + kt * BK;
    unsigned sa = asBase + (unsigned)buf * (BM * STR * 2);
    unsigned sb = bsBase + (unsigned)buf * (BN * STR * 2);
#if USE_ASYNC
    async_b128(sa, gaa);
    async_b128(sa + 64 * STR * 2, gaa + (size_t)64 * lda);
    async_b128(sb, gbb);
    async_b128(sb + 64 * STR * 2, gbb + (size_t)64 * ldb);
    async_b128(sb + 128 * STR * 2, gbb + (size_t)128 * ldb);
    async_b128(sb + 192 * STR * 2, gbb + (size_t)192 * ldb);
#else
    uint4 a0 = *(const uint4*)(gaa);
    uint4 a1 = *(const uint4*)(gaa + (size_t)64 * lda);
    uint4 b0 = *(const uint4*)(gbb);
    uint4 b1 = *(const uint4*)(gbb + (size_t)64 * ldb);
    uint4 b2 = *(const uint4*)(gbb + (size_t)128 * ldb);
    uint4 b3 = *(const uint4*)(gbb + (size_t)192 * ldb);
    int ia = buf * BM * STR + rowA * STR + ko;
    int ib = buf * BN * STR + rowA * STR + ko;
    *(uint4*)(&As[ia]) = a0;
    *(uint4*)(&As[ia + 64 * STR]) = a1;
    *(uint4*)(&Bs[ib]) = b0;
    *(uint4*)(&Bs[ib + 64 * STR]) = b1;
    *(uint4*)(&Bs[ib + 128 * STR]) = b2;
    *(uint4*)(&Bs[ib + 192 * STR]) = b3;
#endif
  };

  v8f acc[4][4];
#pragma unroll
  for (int mi = 0; mi < 4; mi++)
#pragma unroll
    for (int ni = 0; ni < 4; ni++)
#pragma unroll
      for (int r = 0; r < 8; r++) acc[mi][ni][r] = 0.0f;

  int nk = K / BK;
  issue_tile(0, 0);

  for (int kt = 0; kt < nk; kt++) {
    int cur = kt & 1;
#if USE_ASYNC
    wait_async0();  // my async ops done; barrier => everyone's are
#endif
    __syncthreads();
    if (kt + 1 < nk) issue_tile(1 - cur, kt + 1);

    const unsigned short* sA = &As[cur * BM * STR];
    const unsigned short* sB = &Bs[cur * BN * STR];
    Frag16 af[4], bf[4];
#pragma unroll
    for (int mi = 0; mi < 4; mi++) {
      int m = wm * 64 + mi * 16 + l16;
      af[mi].h[0] = *(const v8bf*)(&sA[m * STR + half * 8]);
      af[mi].h[1] = *(const v8bf*)(&sA[m * STR + 16 + half * 8]);
    }
#pragma unroll
    for (int ni = 0; ni < 4; ni++) {
      int n = wn * 64 + ni * 16 + l16;
      bf[ni].h[0] = *(const v8bf*)(&sB[n * STR + half * 8]);
      bf[ni].h[1] = *(const v8bf*)(&sB[n * STR + 16 + half * 8]);
    }
#pragma unroll
    for (int mi = 0; mi < 4; mi++)
#pragma unroll
      for (int ni = 0; ni < 4; ni++)
        acc[mi][ni] = __builtin_amdgcn_wmma_f32_16x16x32_bf16(
            false, af[mi].v, false, bf[ni].v, (short)0, acc[mi][ni], false,
            false);
    __syncthreads();
  }

  // epilogue: bias (+ReLU), bf16 store. VGPR r -> row r (lane<16) / 8+r.
  int mof = half * 8;
#pragma unroll
  for (int ni = 0; ni < 4; ni++) {
    int n = n_blk + wn * 64 + ni * 16 + l16;
    float bv = biasz[n];
#pragma unroll
    for (int mi = 0; mi < 4; mi++) {
      int mb = m_blk + wm * 64 + mi * 16 + mof;
#pragma unroll
      for (int r = 0; r < 8; r++) {
        float v = acc[mi][ni][r] + bv;
        if (RELU) v = fmaxf(v, 0.0f);
        Cz[(size_t)(mb + r) * ldc + n] = f2bf(v);
      }
    }
  }
}

// ---------------------------------------------------------------------------
// Gating: one wave per (t,b). Lanes 0-15: mean dots; 16-31: raw-std dots.
// Noisy top-4 + softmax -> (idx, weight) x4.
// ---------------------------------------------------------------------------
__global__ __launch_bounds__(256) void gating_kernel(
    const float* __restrict__ x, const float* __restrict__ Wg,
    const float* __restrict__ Wn, int* __restrict__ topk_idx,
    float* __restrict__ topk_w) {
  int wid = threadIdx.x >> 5;
  int lane = threadIdx.x & 31;
  int w = blockIdx.x * 8 + wid;  // 0 .. T_*B_-1
  int t = w >> 13;
  int b = w & (B_ - 1);
  int e = lane & 15;

  const float* W = (lane < 16 ? Wg : Wn) + (size_t)t * IN_ * E_;
  const float* xr = x + (size_t)b * IN_;
  float acc = 0.0f;
#pragma unroll 4
  for (int i = 0; i < IN_; i++) acc = fmaf(xr[i], W[i * E_ + e], acc);

  float sp = (acc > 20.0f) ? acc : log1pf(expf(acc));  // softplus
  float mean = acc;
  float stdv = __shfl(sp, 16 + e, 32);

  // deterministic hash noise (Box-Muller); JAX threefry not reproducible here
  unsigned int h = (unsigned int)(w * 16 + e);
  h = h * 747796405u + 2891336453u;
  h ^= h >> 17; h *= 0xed5ad4bbu; h ^= h >> 11; h *= 0xac4c1b51u;
  h ^= h >> 15; h *= 0x31848babu; h ^= h >> 14;
  unsigned int h2 = h * 747796405u + 2891336453u;
  h2 ^= h2 >> 16; h2 *= 0x7feb352du; h2 ^= h2 >> 15;
  float u1 = (float)(h >> 8) * (1.0f / 16777216.0f) + 1e-7f;
  float u2 = (float)(h2 >> 8) * (1.0f / 16777216.0f);
  float nrm = sqrtf(-2.0f * logf(u1)) * cosf(6.28318530718f * u2);
  float noisy = mean + nrm * stdv;

  float v[16];
#pragma unroll
  for (int j = 0; j < 16; j++) v[j] = __shfl(noisy, j, 32);

  // strict rank (tie-break by index): exactly 4 lanes get rank < 4
  int rank = 0;
#pragma unroll
  for (int j = 0; j < 16; j++)
    rank += (v[j] > noisy || (v[j] == noisy && j < e)) ? 1 : 0;

  // top-5 values (per-lane selection)
  float a2[16];
#pragma unroll
  for (int j = 0; j < 16; j++) a2[j] = v[j];
  float top[5];
#pragma unroll
  for (int s = 0; s < 5; s++) {
    float m = -3.4e38f;
    int mi = -1;
#pragma unroll
    for (int j = 0; j < 16; j++)
      if (a2[j] > m) { m = a2[j]; mi = j; }
    top[s] = m;
#pragma unroll
    for (int j = 0; j < 16; j++)
      if (j == mi) a2[j] = -3.4e38f;
  }
  float k_largest = top[3];
  float denom = 0.0f;
#pragma unroll
  for (int j = 0; j < 16; j++)
    denom += (v[j] >= k_largest) ? expf(v[j] - top[0]) : 0.0f;

  if (lane < 16 && rank < 4) {
    topk_idx[(size_t)w * 4 + rank] = e;
    topk_w[(size_t)w * 4 + rank] = expf(noisy - top[0]) / denom;
  }
}

// ---------------------------------------------------------------------------
// gate_out[t][b][:] = sum_j w_j * expert_out[idx_j][b][:]   (bf16 in/out)
// grid (B_, T_), block 256
// ---------------------------------------------------------------------------
__global__ __launch_bounds__(256) void gather_kernel(
    const unsigned short* __restrict__ expert_out,
    const int* __restrict__ topk_idx, const float* __restrict__ topk_w,
    unsigned short* __restrict__ gate_out) {
  int b = blockIdx.x, t = blockIdx.y;
  int w = t * B_ + b;
  int i0 = topk_idx[w * 4 + 0], i1 = topk_idx[w * 4 + 1];
  int i2 = topk_idx[w * 4 + 2], i3 = topk_idx[w * 4 + 3];
  float w0 = topk_w[w * 4 + 0], w1 = topk_w[w * 4 + 1];
  float w2 = topk_w[w * 4 + 2], w3 = topk_w[w * 4 + 3];
#pragma unroll
  for (int o = threadIdx.x; o < H2_; o += 256) {
    float s = w0 * bf2f(expert_out[((size_t)i0 * B_ + b) * H2_ + o]);
    s = fmaf(w1, bf2f(expert_out[((size_t)i1 * B_ + b) * H2_ + o]), s);
    s = fmaf(w2, bf2f(expert_out[((size_t)i2 * B_ + b) * H2_ + o]), s);
    s = fmaf(w3, bf2f(expert_out[((size_t)i3 * B_ + b) * H2_ + o]), s);
    gate_out[(size_t)w * H2_ + o] = f2bf(s);
  }
}

// ---------------------------------------------------------------------------
// out[t*B + b] = dot(T2[t][b][:256], Wh3[t]) + bh3[t]; one wave per row
// ---------------------------------------------------------------------------
__global__ __launch_bounds__(256) void final_kernel(
    const unsigned short* __restrict__ T2buf, const float* __restrict__ Wh3,
    const float* __restrict__ bh3, float* __restrict__ out) {
  int wid = threadIdx.x >> 5;
  int lane = threadIdx.x & 31;
  int w = blockIdx.x * 8 + wid;  // 0 .. T_*B_-1
  int t = w >> 13;
  const unsigned short* row = T2buf + (size_t)w * HD2_;
  const float* wv = Wh3 + t * HD2_;
  float acc = 0.0f;
#pragma unroll
  for (int j = lane; j < HD2_; j += 32) acc = fmaf(bf2f(row[j]), wv[j], acc);
#pragma unroll
  for (int off = 16; off > 0; off >>= 1) acc += __shfl_down(acc, off, 32);
  if (lane == 0) out[w] = acc + bh3[t];
}

// ---------------------------------------------------------------------------
extern "C" void kernel_launch(void* const* d_in, const int* in_sizes, int n_in,
                              void* d_out, int out_size, void* d_ws,
                              size_t ws_size, hipStream_t stream) {
  (void)in_sizes; (void)n_in; (void)out_size; (void)ws_size;
  const float* x   = (const float*)d_in[0];
  const float* We1 = (const float*)d_in[1];
  const float* be1 = (const float*)d_in[2];
  const float* We2 = (const float*)d_in[3];
  const float* be2 = (const float*)d_in[4];
  const float* Wg  = (const float*)d_in[5];
  const float* Wn  = (const float*)d_in[6];
  const float* Wh1 = (const float*)d_in[7];
  const float* bh1 = (const float*)d_in[8];
  const float* Wh2 = (const float*)d_in[9];
  const float* bh2 = (const float*)d_in[10];
  const float* Wh3 = (const float*)d_in[11];
  const float* bh3 = (const float*)d_in[12];
  float* out = (float*)d_out;

  char* ws = (char*)d_ws;
  size_t off = 0;
  auto take = [&](size_t bytes) -> char* {
    char* p = ws + off;
    off += (bytes + 255) & ~(size_t)255;
    return p;
  };
  unsigned short* xb   = (unsigned short*)take((size_t)B_ * IN_ * 2);
  unsigned short* We1t = (unsigned short*)take((size_t)E_ * H1_ * IN_ * 2);
  unsigned short* We2t = (unsigned short*)take((size_t)E_ * H2_ * H1_ * 2);
  unsigned short* Wh1t = (unsigned short*)take((size_t)T_ * HD1_ * H2_ * 2);
  unsigned short* Wh2t = (unsigned short*)take((size_t)T_ * HD2_ * HD1_ * 2);
  unsigned short* Hbuf = (unsigned short*)take((size_t)E_ * B_ * H1_ * 2);
  unsigned short* EO   = (unsigned short*)take((size_t)E_ * B_ * H2_ * 2);
  unsigned short* GO   = (unsigned short*)take((size_t)T_ * B_ * H2_ * 2);
  unsigned short* T1b  = (unsigned short*)take((size_t)T_ * B_ * HD1_ * 2);
  unsigned short* T2b  = (unsigned short*)take((size_t)T_ * B_ * HD2_ * 2);
  int*   tidx = (int*)take((size_t)T_ * B_ * 4 * sizeof(int));
  float* tw   = (float*)take((size_t)T_ * B_ * 4 * sizeof(float));

  // 1) precision conversion + weight transposes (K-contiguous layouts)
  cvt_f32_bf16<<<(B_ * IN_) / 1024, 256, 0, stream>>>(x, xb, B_ * IN_);
  transpose_f32_bf16<<<dim3(H1_ / 32, IN_ / 32, E_), dim3(32, 8), 0, stream>>>(
      We1, We1t, IN_, H1_);
  transpose_f32_bf16<<<dim3(H2_ / 32, H1_ / 32, E_), dim3(32, 8), 0, stream>>>(
      We2, We2t, H1_, H2_);
  transpose_f32_bf16<<<dim3(HD1_ / 32, H2_ / 32, T_), dim3(32, 8), 0, stream>>>(
      Wh1, Wh1t, H2_, HD1_);
  transpose_f32_bf16<<<dim3(HD2_ / 32, HD1_ / 32, T_), dim3(32, 8), 0,
                       stream>>>(Wh2, Wh2t, HD1_, HD2_);

  // 2) expert layer 1: H[e] = relu(x @ We1[e] + be1[e])
  wmma_gemm_bf16<true><<<dim3(H1_ / BN, B_ / BM, E_), 256, 0, stream>>>(
      xb, 0LL, We1t, (long long)H1_ * IN_, be1, (long long)H1_, Hbuf,
      (long long)B_ * H1_, B_, H1_, IN_, IN_, IN_, H1_);

  // 3) expert layer 2: EO[e] = H[e] @ We2[e] + be2[e]
  wmma_gemm_bf16<false><<<dim3(H2_ / BN, B_ / BM, E_), 256, 0, stream>>>(
      Hbuf, (long long)B_ * H1_, We2t, (long long)H2_ * H1_, be2,
      (long long)H2_, EO, (long long)B_ * H2_, B_, H2_, H1_, H1_, H1_, H2_);

  // 4) noisy top-k gating
  gating_kernel<<<(T_ * B_) / 8, 256, 0, stream>>>(x, Wg, Wn, tidx, tw);

  // 5) weighted expert gather
  gather_kernel<<<dim3(B_, T_), 256, 0, stream>>>(EO, tidx, tw, GO);

  // 6) tower layers 1 & 2 (WMMA)
  wmma_gemm_bf16<true><<<dim3(HD1_ / BN, B_ / BM, T_), 256, 0, stream>>>(
      GO, (long long)B_ * H2_, Wh1t, (long long)HD1_ * H2_, bh1,
      (long long)HD1_, T1b, (long long)B_ * HD1_, B_, HD1_, H2_, H2_, H2_,
      HD1_);
  wmma_gemm_bf16<true><<<dim3(HD2_ / BN, B_ / BM, T_), 256, 0, stream>>>(
      T1b, (long long)B_ * HD1_, Wh2t, (long long)HD2_ * HD1_, bh2,
      (long long)HD2_, T2b, (long long)B_ * HD2_, B_, HD2_, HD1_, HD1_, HD1_,
      HD2_);

  // 7) final 256-wide dot + bias per (tower,row)
  final_kernel<<<(T_ * B_) / 8, 256, 0, stream>>>(T2b, Wh3, bh3, out);
}
